// MyTraceAnomalyModel_15393162789543
// MI455X (gfx1250) — matrix operand order repaired and checked by reading, not compiled, not run on hardware
//
#include <hip/hip_runtime.h>
#include <hip/hip_bf16.h>

#define N_NODES 500000
#define N_EDGES 1000000
#define EMB     64
#define FDIM    128
#define L_DIM   8
#define OP_CNT_ 5000
#define IN_DIM  200          // 3*EMB + L
#define K1T     7            // ceil(224/32) k-tiles for layer 1
#define FSTB    232          // feats LDS row stride (bf16): mult of 8, conflict-free
#define HSTB    136          // h LDS row stride (bf16)
#define ZST     136          // z/agg/yd LDS row stride (bf16)

typedef __bf16 bf16_t;
typedef __attribute__((ext_vector_type(16))) __bf16 v16bf;
typedef __attribute__((ext_vector_type(8)))  __bf16 v8bf;
typedef __attribute__((ext_vector_type(8)))  float  v8f;

// ---------------------------------------------------------------- WMMA helpers
static __device__ inline v8f wmma_bf16(v16bf a, v16bf b, v8f c) {
  // D = A(16x32 bf16) * B(32x16 bf16) + C(16x16 f32)
  return __builtin_amdgcn_wmma_f32_16x16x32_bf16(false, a, false, b, (short)0, c,
                                                 false, false);
}

// A fragment (16x32 bf16) from a bf16 LDS tile (row-major, stride mult-of-8).
// Lane layout: m=lane&15 row, h=lane>>4; elements 0..7 = K (k0+8h)..(k0+8h+7),
// elements 8..15 = K (k0+16+8h)..(k0+16+8h+7): two contiguous 16B chunks.
static __device__ inline v16bf load_a_lds(const bf16_t* t, int stride, int k0, int lane) {
  int m = lane & 15, h = lane >> 4;
  const bf16_t* r = t + m * stride + k0 + 8 * h;
  v8bf lo = *(const v8bf*)(r);
  v8bf hi = *(const v8bf*)(r + 16);
  return __builtin_shufflevector(lo, hi, 0, 1, 2, 3, 4, 5, 6, 7,
                                 8, 9, 10, 11, 12, 13, 14, 15);
}

// B fragment from pre-packed weights: fragment (kt,nt) for `lane` is 16
// contiguous bf16 (32B) at P[((kt*NT+nt)*32+lane)*16]  -> coalesced b128 loads.
static __device__ inline v16bf load_b_packed(const bf16_t* P, int NT, int kt, int nt,
                                             int lane) {
  return *(const v16bf*)(P + (((size_t)kt * NT + nt) * 32 + lane) * 16);
}

// ---------------------------------------------------------------- pack kernel
// Convert W[K][ncol] (f32, row-major) into B-fragment-swizzled bf16, with
// zero padding for k>=kvalid / n>=nvalid. One thread per (kt,nt,lane) fragment.
__global__ void pack_w_kernel(const float* __restrict__ W, bf16_t* __restrict__ P,
                              int kvalid, int nvalid, int ncol, int KT, int NT) {
  int t = blockIdx.x * blockDim.x + threadIdx.x;
  if (t >= KT * NT * 32) return;
  int lane = t & 31;
  int nt = (t >> 5) % NT;
  int kt = (t >> 5) / NT;
  int n = nt * 16 + (lane & 15);
  int h = lane >> 4;
  v16bf f;
#pragma unroll
  for (int j = 0; j < 8; ++j) {
    int k = kt * 32 + 8 * h + 2 * j + (j >= 4 ? 8 : 0);
    f[2 * j]     = (__bf16)((n < nvalid && k     < kvalid) ? W[(size_t)k * ncol + n] : 0.0f);
    f[2 * j + 1] = (__bf16)((n < nvalid && k + 1 < kvalid) ? W[(size_t)(k + 1) * ncol + n] : 0.0f);
  }
  *(v16bf*)(P + (size_t)t * 16) = f;
}

// ---------------------------------------------------------------- kernels
__global__ void zero_kernel(float* p, size_t n) {
  size_t i = (size_t)blockIdx.x * blockDim.x + threadIdx.x;
  size_t step = (size_t)gridDim.x * blockDim.x;
  for (; i < n; i += step) p[i] = 0.0f;
}

// Encoder: feats = [op_e|sv_e|st_e|lat]; h = relu(feats@W1+b1); y = h@W2+b2;
// z = mu + exp(0.5*tanh(lv))*eps; KL partial accumulated.
__global__ __launch_bounds__(128) void encode_kernel(
    const int* __restrict__ op_id, const int* __restrict__ sv_id,
    const int* __restrict__ st_id, const float* __restrict__ lat,
    const float* __restrict__ eps,
    const float* __restrict__ op_e, const float* __restrict__ sv_e,
    const float* __restrict__ st_e,
    const bf16_t* __restrict__ W1p, const float* __restrict__ b1,
    const bf16_t* __restrict__ W2p, const float* __restrict__ b2,
    float* __restrict__ zg, float* __restrict__ accum) {
  __shared__ bf16_t sm[4 * 16 * FSTB];        // 29696 B
  const int lane = threadIdx.x & 31, wid = threadIdx.x >> 5;
  const int r0 = blockIdx.x * 64 + wid * 16;
  bf16_t* fl = sm + wid * 16 * FSTB;

  // gather feats tile as bf16 (zero-padded cols 200..231)
  for (int idx = lane; idx < 16 * FSTB; idx += 32) {
    int row = idx / FSTB, col = idx - row * FSTB;
    int node = min(r0 + row, N_NODES - 1);
    float v = 0.0f;
    if (col < 64)        v = op_e[(size_t)op_id[node] * EMB + col];
    else if (col < 128)  v = sv_e[(size_t)sv_id[node] * EMB + (col - 64)];
    else if (col < 192)  v = st_e[(size_t)st_id[node] * EMB + (col - 128)];
    else if (col < 200)  v = lat[(size_t)node * L_DIM + (col - 192)];
    fl[idx] = (__bf16)v;
  }
  __syncthreads();

  const v8f vz = {0.f, 0.f, 0.f, 0.f, 0.f, 0.f, 0.f, 0.f};
  v8f acc[8];
#pragma unroll
  for (int nt = 0; nt < 8; ++nt) acc[nt] = vz;

  // layer 1: 16x128 tile, K = 224 (padded in pack)
#pragma unroll
  for (int kt = 0; kt < K1T; ++kt) {
    v16bf a = load_a_lds(fl, FSTB, kt * 32, lane);
#pragma unroll
    for (int nt = 0; nt < 8; ++nt)
      acc[nt] = wmma_bf16(a, load_b_packed(W1p, 8, kt, nt, lane), acc[nt]);
  }

  // relu + bias -> h tile (bf16) back into (same) LDS region, stride HSTB
  const int m0 = (lane >> 4) * 8, nl = lane & 15;
  __syncthreads();
#pragma unroll
  for (int nt = 0; nt < 8; ++nt) {
#pragma unroll
    for (int j = 0; j < 8; ++j) {
      int col = nt * 16 + nl;
      fl[(m0 + j) * HSTB + col] = (__bf16)fmaxf(acc[nt][j] + b1[col], 0.0f);
    }
  }
  __syncthreads();

  v16bf ha[4];
#pragma unroll
  for (int kk = 0; kk < 4; ++kk) ha[kk] = load_a_lds(fl, HSTB, kk * 32, lane);

  // layer 2: paired (mu, lv) column tiles, reparam, KL partial
  float kl = 0.0f;
#pragma unroll
  for (int nt = 0; nt < 8; ++nt) {
    v8f amu = vz, alv = vz;
#pragma unroll
    for (int kk = 0; kk < 4; ++kk) {
      amu = wmma_bf16(ha[kk], load_b_packed(W2p, 16, kk, nt,     lane), amu);
      alv = wmma_bf16(ha[kk], load_b_packed(W2p, 16, kk, nt + 8, lane), alv);
    }
#pragma unroll
    for (int j = 0; j < 8; ++j) {
      int node = r0 + m0 + j;
      int col = nt * 16 + nl;
      float zmu = amu[j] + b2[col];
      float zlv = tanhf(alv[j] + b2[FDIM + col]);
      if (node < N_NODES) {
        float e = eps[(size_t)node * FDIM + col];
        zg[(size_t)node * FDIM + col] = zmu + __expf(0.5f * zlv) * e;
        kl += zlv + 1.0f - __expf(zlv) - zmu * zmu;
      }
    }
  }
#pragma unroll
  for (int off = 16; off >= 1; off >>= 1) kl += __shfl_xor(kl, off, 32);
  if (lane == 0) atomicAdd(accum + 0, kl);
}

// agg[dst] += z[src] : 32 lanes per edge, float4 per lane, global f32 atomics.
__global__ void scatter_kernel(const int* __restrict__ src, const int* __restrict__ dst,
                               const float* __restrict__ z, float* __restrict__ agg) {
  long long t = (long long)blockIdx.x * blockDim.x + threadIdx.x;
  if (t >= (long long)N_EDGES * 32) return;
  int e = (int)(t >> 5);
  int q = ((int)t & 31) * 4;
  int s = src[e], d = dst[e];
  float4 v = *(const float4*)(z + (size_t)s * FDIM + q);
  float* p = agg + (size_t)d * FDIM + q;
  atomicAdd(p + 0, v.x);
  atomicAdd(p + 1, v.y);
  atomicAdd(p + 2, v.z);
  atomicAdd(p + 3, v.w);
}

// Decoder: yd = relu(z@Wself + agg@Wagg); mu = yd@muW+mub; lv = yd@lvW+lvb
__global__ __launch_bounds__(128) void decode_kernel(
    const float* __restrict__ zg, const float* __restrict__ ag,
    const bf16_t* __restrict__ Wsp, const bf16_t* __restrict__ Wap,
    const bf16_t* __restrict__ muWp, const float* __restrict__ mub,
    const bf16_t* __restrict__ lvWp, const float* __restrict__ lvb,
    float* __restrict__ muo, float* __restrict__ lvo) {
  __shared__ bf16_t sm[4 * 2 * 16 * ZST];     // 34816 B
  const int lane = threadIdx.x & 31, wid = threadIdx.x >> 5;
  const int r0 = blockIdx.x * 64 + wid * 16;
  bf16_t* zt = sm + wid * (2 * 16 * ZST);
  bf16_t* at = zt + 16 * ZST;

  for (int idx = lane; idx < 16 * FDIM; idx += 32) {
    int row = idx >> 7, col = idx & 127;
    int node = min(r0 + row, N_NODES - 1);
    zt[row * ZST + col] = (__bf16)zg[(size_t)node * FDIM + col];
    at[row * ZST + col] = (__bf16)ag[(size_t)node * FDIM + col];
  }
  __syncthreads();

  v16bf za[4], aa[4];
#pragma unroll
  for (int kk = 0; kk < 4; ++kk) {
    za[kk] = load_a_lds(zt, ZST, kk * 32, lane);
    aa[kk] = load_a_lds(at, ZST, kk * 32, lane);
  }

  const v8f vz = {0.f, 0.f, 0.f, 0.f, 0.f, 0.f, 0.f, 0.f};
  const int m0 = (lane >> 4) * 8, nl = lane & 15;
  __syncthreads();
#pragma unroll
  for (int nt = 0; nt < 8; ++nt) {
    v8f acc = vz;
#pragma unroll
    for (int kk = 0; kk < 4; ++kk) {
      acc = wmma_bf16(za[kk], load_b_packed(Wsp, 8, kk, nt, lane), acc);
      acc = wmma_bf16(aa[kk], load_b_packed(Wap, 8, kk, nt, lane), acc);
    }
#pragma unroll
    for (int j = 0; j < 8; ++j)   // yd (bf16) reuses the z tile region
      zt[(m0 + j) * ZST + nt * 16 + nl] = (__bf16)fmaxf(acc[j], 0.0f);
  }
  __syncthreads();

  v8f amu = vz, alv = vz;
#pragma unroll
  for (int kk = 0; kk < 4; ++kk) {
    v16bf yd = load_a_lds(zt, ZST, kk * 32, lane);
    amu = wmma_bf16(yd, load_b_packed(muWp, 1, kk, 0, lane), amu);
    alv = wmma_bf16(yd, load_b_packed(lvWp, 1, kk, 0, lane), alv);
  }
  if (nl < L_DIM) {
#pragma unroll
    for (int j = 0; j < 8; ++j) {
      int node = r0 + m0 + j;
      if (node < N_NODES) {
        muo[(size_t)node * L_DIM + nl] = amu[j] + mub[nl];
        lvo[(size_t)node * L_DIM + nl] = alv[j] + lvb[nl];
      }
    }
  }
}

// Per-node 8x8 einsum with opwise weights, loss, segment sums.
__global__ void loss_kernel(const int* __restrict__ op_id, const float* __restrict__ lat,
                            const float* __restrict__ opw,
                            const float* __restrict__ muo, const float* __restrict__ lvo,
                            float* __restrict__ op_loss, float* __restrict__ op_cnt,
                            float* __restrict__ accum) {
  int n = blockIdx.x * blockDim.x + threadIdx.x;
  float lm = 0.0f;
  bool valid = n < N_NODES;
  if (valid) {
    int op = op_id[n];
    const float* w = opw + (size_t)op * (2 * L_DIM * L_DIM);   // [8][16]
    float mu[8], lv[8];
#pragma unroll
    for (int l = 0; l < 8; ++l) {
      mu[l] = muo[(size_t)n * L_DIM + l];
      lv[l] = lvo[(size_t)n * L_DIM + l];
    }
#pragma unroll
    for (int k = 0; k < 8; ++k) {
      float m2 = 0.0f, l2 = 0.0f;
#pragma unroll
      for (int l = 0; l < 8; ++l) {
        m2 += mu[l] * w[l * 16 + k];
        l2 += lv[l] * w[l * 16 + 8 + k];
      }
      float d = m2 - lat[(size_t)n * L_DIM + k];
      lm += d * d / (2.0f * __expf(l2) + 1e-7f) + 0.5f * l2;
    }
    lm *= 0.125f;                      // mean over L
    atomicAdd(op_loss + op, lm);
    atomicAdd(op_cnt + op, 1.0f);
  }
  float s = valid ? lm : 0.0f;
#pragma unroll
  for (int off = 16; off >= 1; off >>= 1) s += __shfl_xor(s, off, 32);
  if ((threadIdx.x & 31) == 0) atomicAdd(accum + 1, s);
}

__global__ void finalize_kernel(const float* __restrict__ accum, float* __restrict__ out) {
  // total = mean(loss_node) + (-0.5)*mean(z_lv + 1 - exp(z_lv) - z_mu^2)
  out[0] = accum[1] / (float)N_NODES
         - 0.5f * accum[0] / ((float)N_NODES * (float)FDIM);
}

// ---------------------------------------------------------------- launch
extern "C" void kernel_launch(void* const* d_in, const int* in_sizes, int n_in,
                              void* d_out, int out_size, void* d_ws, size_t ws_size,
                              hipStream_t stream) {
  const int*   op_id = (const int*)d_in[0];
  const int*   sv_id = (const int*)d_in[1];
  const int*   st_id = (const int*)d_in[2];
  const float* lat   = (const float*)d_in[3];
  const int*   esrc  = (const int*)d_in[4];
  const int*   edst  = (const int*)d_in[5];
  const float* eps   = (const float*)d_in[6];
  const float* op_e  = (const float*)d_in[7];
  const float* sv_e  = (const float*)d_in[8];
  const float* st_e  = (const float*)d_in[9];
  const float* W1    = (const float*)d_in[10];
  const float* b1    = (const float*)d_in[11];
  const float* W2    = (const float*)d_in[12];
  const float* b2    = (const float*)d_in[13];
  const float* Wself = (const float*)d_in[14];
  const float* Wagg  = (const float*)d_in[15];
  const float* muW   = (const float*)d_in[16];
  const float* mub   = (const float*)d_in[17];
  const float* lvW   = (const float*)d_in[18];
  const float* lvb   = (const float*)d_in[19];
  const float* opw   = (const float*)d_in[20];

  float* ws = (float*)d_ws;
  size_t off = 0;
  float* zg  = ws + off; off += (size_t)N_NODES * FDIM;   // z [N,128]
  float* ag  = ws + off; off += (size_t)N_NODES * FDIM;   // agg [N,128]
  float* muo = ws + off; off += (size_t)N_NODES * L_DIM;  // mu  [N,8]
  float* lvo = ws + off; off += (size_t)N_NODES * L_DIM;  // lv  [N,8]

  // packed bf16 weights (fragment-swizzled); 32B-aligned region
  bf16_t* pk = (bf16_t*)(ws + off);
  size_t poff = 0;
  bf16_t* W1p  = pk + poff; poff += (size_t)K1T * 8 * 512;   // KT*NT*32*16
  bf16_t* W2p  = pk + poff; poff += (size_t)4 * 16 * 512;
  bf16_t* Wsp  = pk + poff; poff += (size_t)4 * 8 * 512;
  bf16_t* Wap  = pk + poff; poff += (size_t)4 * 8 * 512;
  bf16_t* muWp = pk + poff; poff += (size_t)4 * 1 * 512;
  bf16_t* lvWp = pk + poff; poff += (size_t)4 * 1 * 512;
  float* accum = (float*)(pk + poff);                        // [kl_sum, lat_sum]

  // pack weights into WMMA B-fragment layout (bf16)
  pack_w_kernel<<<(K1T * 8 * 32 + 127) / 128, 128, 0, stream>>>(W1, W1p, IN_DIM, FDIM, FDIM, K1T, 8);
  pack_w_kernel<<<(4 * 16 * 32 + 127) / 128, 128, 0, stream>>>(W2, W2p, FDIM, 2 * FDIM, 2 * FDIM, 4, 16);
  pack_w_kernel<<<(4 * 8 * 32 + 127) / 128, 128, 0, stream>>>(Wself, Wsp, FDIM, FDIM, FDIM, 4, 8);
  pack_w_kernel<<<(4 * 8 * 32 + 127) / 128, 128, 0, stream>>>(Wagg, Wap, FDIM, FDIM, FDIM, 4, 8);
  pack_w_kernel<<<(4 * 1 * 32 + 127) / 128, 128, 0, stream>>>(muW, muWp, FDIM, L_DIM, L_DIM, 4, 1);
  pack_w_kernel<<<(4 * 1 * 32 + 127) / 128, 128, 0, stream>>>(lvW, lvWp, FDIM, L_DIM, L_DIM, 4, 1);

  zero_kernel<<<2048, 256, 0, stream>>>(ag, (size_t)N_NODES * FDIM);
  zero_kernel<<<1,    256, 0, stream>>>(accum, (size_t)2);
  zero_kernel<<<64,   256, 0, stream>>>((float*)d_out, (size_t)out_size);

  float* out     = (float*)d_out;
  float* op_loss = out + 1;
  float* op_cnt  = out + 1 + OP_CNT_;

  int gb = (N_NODES + 63) / 64;   // 64 nodes per block (4 waves x 16-node tiles)
  encode_kernel<<<gb, 128, 0, stream>>>(op_id, sv_id, st_id, lat, eps,
                                        op_e, sv_e, st_e, W1p, b1, W2p, b2,
                                        zg, accum);

  long long sthreads = (long long)N_EDGES * 32;
  scatter_kernel<<<(unsigned)((sthreads + 255) / 256), 256, 0, stream>>>(esrc, edst, zg, ag);

  decode_kernel<<<gb, 128, 0, stream>>>(zg, ag, Wsp, Wap, muWp, mub, lvWp, lvb,
                                        muo, lvo);

  loss_kernel<<<(N_NODES + 255) / 256, 256, 0, stream>>>(op_id, lat, opw, muo, lvo,
                                                         op_loss, op_cnt, accum);

  finalize_kernel<<<1, 1, 0, stream>>>(accum, out);
}
